// MultiHeadAttention_33225867002321
// MI455X (gfx1250) — compile-verified, matrix operand order checked by reading, not run on hardware
//
#include <hip/hip_runtime.h>

// MHA forward for gfx1250 (MI455X): all GEMMs via v_wmma_f32_16x16x32_f16,
// TDM (tensor_load_to_lds) staging of the Q tile, NT stores for the 537MB
// attn_weights stream.  d_out = [output (B*S*D f32)] ++ [attn (B*H*S*S f32)].

#define D_MODEL   1024
#define NUM_HEADS 16
#define HEAD_DIM  64
#define BATCH     2
#define SEQ       2048
#define SLD       2052   // LDS row stride for 2048-wide score stripe (+pad)

typedef _Float16 v16h __attribute__((ext_vector_type(16)));
typedef float    v8f  __attribute__((ext_vector_type(8)));
typedef unsigned int u32x4 __attribute__((ext_vector_type(4)));
typedef int          i32x8 __attribute__((ext_vector_type(8)));
typedef int          i32x4 __attribute__((ext_vector_type(4)));

#define HAVE_TDM (__has_builtin(__builtin_amdgcn_tensor_load_to_lds) && \
                  __has_builtin(__builtin_amdgcn_s_wait_tensorcnt))

__device__ __forceinline__ v8f wmma_f16(v16h a, v16h b, v8f c) {
  // (neg_a, A, neg_b, B, c_mod, C, reuse_a, reuse_b)
  return __builtin_amdgcn_wmma_f32_16x16x32_f16(false, a, false, b, (short)0, c,
                                                false, false);
}

// Fragment: lane holds row m = lane&15 of a row-major [16 x >=32] tile,
// 16 f16 values at K = {g*16 + (lane>>4)*8 + i}.  Used for the A operand
// (row = M) and for the B operand whenever B^T is row-major (row = N,
// K contiguous) -- which after the V-transpose trick is every GEMM here.
template <typename T>
__device__ __forceinline__ v16h load_frag_rowK(const T* __restrict__ p, int ld, int lane) {
  int m  = lane & 15;
  int h8 = (lane >> 4) << 3;
  const T* r = p + (size_t)m * ld;
  v16h a;
#pragma unroll
  for (int g = 0; g < 2; ++g)
#pragma unroll
    for (int i = 0; i < 8; ++i)
      a[g * 8 + i] = (_Float16)r[g * 16 + h8 + i];
  return a;
}

// out[M,N] = scale * (A[M,K] @ W[N,K]^T).  One wave = one 16x16 C tile.
// block = (32,4): 4 waves -> 4 consecutive M tiles, same N tile.
template <typename AT, typename OT, bool NTSTORE>
__global__ __launch_bounds__(128) void gemm_xwT_kernel(
    const AT* __restrict__ A, const float* __restrict__ W, OT* __restrict__ out,
    int M, int N, int K, float scale) {
  int lane  = threadIdx.x & 31;
  int ntile = blockIdx.x;
  int mtile = blockIdx.y * blockDim.y + threadIdx.y;
  if (mtile * 16 >= M) return;  // wave-uniform

  const AT*    Ab = A + (size_t)mtile * 16 * K;
  const float* Wb = W + (size_t)ntile * 16 * K;

  v8f c = {};
  for (int k0 = 0; k0 < K; k0 += 32) {
    v16h a = load_frag_rowK(Ab + k0, K, lane);
    v16h b = load_frag_rowK(Wb + k0, K, lane);
    c = wmma_f16(a, b, c);
  }

  int row = (lane >> 4) * 8;
  int col = lane & 15;
  OT* ob = out + ((size_t)mtile * 16) * N + (size_t)ntile * 16 + col;
#pragma unroll
  for (int r = 0; r < 8; ++r) {
    OT v = (OT)(c[r] * scale);
    if (NTSTORE) __builtin_nontemporal_store(v, ob + (size_t)(row + r) * N);
    else         ob[(size_t)(row + r) * N] = v;
  }
}

// Fused scores -> softmax -> attn_weights write -> context, per (b,h,16-query stripe).
// 256 threads (8 waves). Dynamic LDS: float P[16][SLD] + float pbuf[4*8*32].
__global__ __launch_bounds__(256) void attn_kernel(
    const _Float16* __restrict__ Qh,   // (B,S,D) f16, pre-scaled by 1/sqrt(Dh)
    const _Float16* __restrict__ Kh,   // (B,S,D) f16
    const _Float16* __restrict__ Vt,   // (D_MODEL, B*S) f16  == V transposed
    float* __restrict__ attn_out,      // (B,H,S,S) f32
    _Float16* __restrict__ ctx) {      // (B,S,D) f16
  extern __shared__ float P[];
  float* pbuf = P + 16 * SLD;
  __shared__ float red[16][17];
  __shared__ float rstat[16];
  __shared__ __align__(16) _Float16 qstage[16 * HEAD_DIM];  // 2KB Q tile

  int t    = threadIdx.x;
  int lane = t & 31;
  int w    = t >> 5;  // wave 0..7
  int q0   = blockIdx.x * 16;
  int h    = blockIdx.y;
  int bb   = blockIdx.z;
  int row  = (lane >> 4) * 8;
  int col  = lane & 15;

  const _Float16* Qbase = Qh + ((size_t)bb * SEQ + q0) * D_MODEL + h * HEAD_DIM;

  // ---- Stage Q tile (16x64 f16) into LDS via the Tensor Data Mover ----
#if HAVE_TDM
  if (w == 0) {
    // D# per CDNA5 ISA ch.8: group0 = {count, lds_addr, global_addr, type=2}
    unsigned int lds_byte = (unsigned int)(unsigned long long)(const void*)qstage;
    unsigned long long ga = (unsigned long long)(const void*)Qbase;
    u32x4 g0 = { 1u,                                   // count=1, user mode
                 lds_byte,                             // lds_addr
                 (unsigned int)ga,                     // global_addr[31:0]
                 ((unsigned int)(ga >> 32) & 0x01FFFFFFu) | 0x80000000u };  // [56:32] | type=2
    // group1: data_size=2B, tensor_dim0=64, tensor_dim1=16, tile 64x16,
    //         tensor_dim0_stride=D_MODEL
    i32x8 g1 = { 0x00010000,            // data_size=1 (2 bytes)
                 64 << 16,              // tensor_dim0[15:0]
                 16 << 16,              // tensor_dim0[31:16]=0 | tensor_dim1[15:0]=16
                 64 << 16,              // tensor_dim1[31:16]=0 | tile_dim0=64
                 16,                    // tile_dim1=16, tile_dim2=0
                 D_MODEL,               // tensor_dim0_stride[31:0]
                 0, 0 };                // stride0[47:32]=0, tensor_dim1_stride=0
    i32x4 gz4 = { 0, 0, 0, 0 };         // zero groups 2/3: valid for 2D tensor
    i32x8 gz8 = { 0, 0, 0, 0, 0, 0, 0, 0 };
    __builtin_amdgcn_tensor_load_to_lds(g0, g1, gz4, gz4, gz8, 0);
    __builtin_amdgcn_s_wait_tensorcnt(0);
  }
  __syncthreads();
  v16h a0 = load_frag_rowK(qstage,      HEAD_DIM, lane);
  v16h a1 = load_frag_rowK(qstage + 32, HEAD_DIM, lane);
#else
  v16h a0 = load_frag_rowK(Qbase,      D_MODEL, lane);
  v16h a1 = load_frag_rowK(Qbase + 32, D_MODEL, lane);
#endif

  // ---- Phase 1: scores stripe S[16 x 2048] = Q_tile(16x64) @ K^T ----
  for (int kt = w * 16; kt < (w + 1) * 16; ++kt) {
    const _Float16* Kbase = Kh + ((size_t)bb * SEQ + kt * 16) * D_MODEL + h * HEAD_DIM;
    v16h b0 = load_frag_rowK(Kbase,      D_MODEL, lane);
    v16h b1 = load_frag_rowK(Kbase + 32, D_MODEL, lane);
    v8f c = {};
    c = wmma_f16(a0, b0, c);
    c = wmma_f16(a1, b1, c);
#pragma unroll
    for (int r = 0; r < 8; ++r)
      P[(row + r) * SLD + kt * 16 + col] = c[r];
  }
  __syncthreads();

  // ---- Phase 2: softmax over each of the 16 rows (16 threads per row) ----
  int r16 = t >> 4, j = t & 15;
  float mx = -3.0e38f;
  for (int cix = j; cix < SEQ; cix += 16) mx = fmaxf(mx, P[r16 * SLD + cix]);
  red[r16][j] = mx;
  __syncthreads();
  if (j == 0) {
    float mm = red[r16][0];
#pragma unroll
    for (int i = 1; i < 16; ++i) mm = fmaxf(mm, red[r16][i]);
    rstat[r16] = mm;
  }
  __syncthreads();
  float rowmax = rstat[r16];
  float s = 0.f;
  for (int cix = j; cix < SEQ; cix += 16) {
    float e = __expf(P[r16 * SLD + cix] - rowmax);
    P[r16 * SLD + cix] = e;
    s += e;
  }
  __syncthreads();
  red[r16][j] = s;
  __syncthreads();
  if (j == 0) {
    float tot = red[r16][0];
#pragma unroll
    for (int i = 1; i < 16; ++i) tot += red[r16][i];
    rstat[r16] = 1.0f / tot;
  }
  __syncthreads();

  // normalize in LDS + coalesced NON-TEMPORAL f32 write of attn_weights
  size_t abase = (((size_t)bb * NUM_HEADS + h) * SEQ + q0) * SEQ;
  for (int idx = t; idx < 16 * SEQ; idx += 256) {
    int rr = idx >> 11, cc = idx & (SEQ - 1);
    float p = P[rr * SLD + cc] * rstat[rr];
    P[rr * SLD + cc] = p;
    __builtin_nontemporal_store(p, attn_out + abase + (size_t)rr * SEQ + cc);
  }
  __syncthreads();

  // ---- Phase 3: context(16x64) = P(16x2048) @ V ; V^T is row-major so both
  //      operands are contiguous-K b128 loads.  K split across wave pairs. ----
  int dtile = w & 3, khalf = w >> 2;
  const _Float16* Vtb =
      Vt + (size_t)(h * HEAD_DIM + dtile * 16) * (BATCH * SEQ) + (size_t)bb * SEQ;
  v8f c = {};
  int kbeg = khalf * (SEQ / 2);
  for (int k0 = kbeg; k0 < kbeg + SEQ / 2; k0 += 32) {
    v16h a;
    {
      int mm_ = lane & 15;
      int h8  = (lane >> 4) << 3;
      const float* pr = P + mm_ * SLD + k0;
#pragma unroll
      for (int g = 0; g < 2; ++g)
#pragma unroll
        for (int i = 0; i < 8; ++i)
          a[g * 8 + i] = (_Float16)pr[g * 16 + h8 + i];
    }
    v16h b = load_frag_rowK(Vtb + k0, BATCH * SEQ, lane);  // two b128 loads
    c = wmma_f16(a, b, c);
  }
  if (khalf == 1) {
#pragma unroll
    for (int r = 0; r < 8; ++r) pbuf[(dtile * 8 + r) * 32 + lane] = c[r];
  }
  __syncthreads();
  if (khalf == 0) {
#pragma unroll
    for (int r = 0; r < 8; ++r) {
      float v = c[r] + pbuf[(dtile * 8 + r) * 32 + lane];
      ctx[((size_t)bb * SEQ + q0 + row + r) * D_MODEL + h * HEAD_DIM + dtile * 16 + col] =
          (_Float16)v;
    }
  }
}

extern "C" void kernel_launch(void* const* d_in, const int* in_sizes, int n_in,
                              void* d_out, int out_size, void* d_ws, size_t ws_size,
                              hipStream_t stream) {
  (void)in_sizes; (void)n_in; (void)out_size; (void)ws_size;
  const float* q  = (const float*)d_in[0];
  const float* k  = (const float*)d_in[1];
  const float* v  = (const float*)d_in[2];
  const float* Wq = (const float*)d_in[3];
  const float* Wk = (const float*)d_in[4];
  const float* Wv = (const float*)d_in[5];
  const float* Wo = (const float*)d_in[6];

  float* out  = (float*)d_out;
  float* attn = out + (size_t)BATCH * SEQ * D_MODEL;

  const size_t NTOK = (size_t)BATCH * SEQ;  // 4096 tokens
  _Float16* hQ   = (_Float16*)d_ws;
  _Float16* hK   = hQ + NTOK * D_MODEL;
  _Float16* hVt  = hK + NTOK * D_MODEL;     // (D_MODEL, NTOK): V transposed
  _Float16* hCtx = hVt + NTOK * D_MODEL;    // total ws use: 33.5 MB

  dim3 gblk(32, 4);
  dim3 ggrd(D_MODEL / 16, (int)(NTOK / 64));

  // Projections (fp32 in, f16 out); fold 1/sqrt(HEAD_DIM)=0.125 into Q.
  gemm_xwT_kernel<float, _Float16, false><<<ggrd, gblk, 0, stream>>>(
      q, Wq, hQ, (int)NTOK, D_MODEL, D_MODEL, 0.125f);
  gemm_xwT_kernel<float, _Float16, false><<<ggrd, gblk, 0, stream>>>(
      k, Wk, hK, (int)NTOK, D_MODEL, D_MODEL, 1.0f);
  // V projection with swapped operands: hVt[d, token] = sum_k Wv[d,k] * v[token,k]
  dim3 ggrdV((int)(NTOK / 16), D_MODEL / 64);
  gemm_xwT_kernel<float, _Float16, false><<<ggrdV, gblk, 0, stream>>>(
      Wv, v, hVt, D_MODEL, (int)NTOK, D_MODEL, 1.0f);

  // Fused attention: TDM Q stage -> scores -> softmax -> attn (NT) -> context.
  size_t smem = (size_t)(16 * SLD + 4 * 8 * 32) * sizeof(float);  // ~135 KB
  (void)hipFuncSetAttribute((const void*)attn_kernel,
                            hipFuncAttributeMaxDynamicSharedMemorySize, (int)smem);
  attn_kernel<<<dim3(SEQ / 16, NUM_HEADS, BATCH), 256, smem, stream>>>(
      hQ, hK, hVt, attn, hCtx);

  // output = context @ Wo^T (f16 A, fp32 W, fp32 NT out)
  gemm_xwT_kernel<_Float16, float, true><<<ggrd, gblk, 0, stream>>>(
      hCtx, Wo, out, (int)NTOK, D_MODEL, D_MODEL, 1.0f);
}
